// Decoder_36249523978345
// MI455X (gfx1250) — matrix operand order repaired and checked by reading, not compiled, not run on hardware
//
#include <hip/hip_runtime.h>
#include <hip/hip_bf16.h>

typedef __bf16 bf16;
typedef __attribute__((ext_vector_type(16))) __bf16 v16bf;
typedef __attribute__((ext_vector_type(8)))  float  v8f;

#define B_   512
#define S_   400
#define V_   50000
#define E_   512
#define H_   256
#define EMB_ 128
#define NG_  1024   // 4*H
#define XK_  640    // E + EMB

// ---------------- WMMA helpers (CDNA5 16x16x32 bf16) ----------------

__device__ __forceinline__ v8f wmma_bf16(v16bf a, v16bf b, v8f c) {
  // 8 args: (neg_a, A, neg_b, B, c_mod, C, reuse_a, reuse_b)
  return __builtin_amdgcn_wmma_f32_16x16x32_bf16(false, a, false, b, (short)0, c,
                                                 false, false);
}

// A-matrix 16x32 bf16 fragment: lanes 0-15: M=lane, K = k0..k0+15
// lanes 16-31: M=lane-16, K = k0+16..k0+31
__device__ __forceinline__ v16bf frag_a(const bf16* p, int ld, int outer_base, int k0) {
  int lane = threadIdx.x & 31;
  int half = lane >> 4;
  const bf16* r = p + (size_t)(outer_base + (lane & 15)) * ld + k0 + half * 16;
  v16bf v;
#pragma unroll
  for (int i = 0; i < 16; ++i) v[i] = r[i];
  return v;
}

// B-matrix 32x16 bf16 fragment: lanes 0-15: N=lane, K=k0..k0+15 ; lanes 16-31: K=k0+16..k0+31.
// p is the weight matrix stored row-major as (N x K): B[k][n] = p[n][k].
__device__ __forceinline__ v16bf frag_b(const bf16* p, int ld, int outer_base, int k0) {
  int lane = threadIdx.x & 31;
  int half = lane >> 4;
  const bf16* r = p + (size_t)(outer_base + (lane & 15)) * ld + k0 + half * 16;
  v16bf v;
#pragma unroll
  for (int i = 0; i < 16; ++i) v[i] = r[i];
  return v;
}

__device__ __forceinline__ float sigf(float x) { return 1.0f / (1.0f + __expf(-x)); }

// ---------------- elementwise conversion fp32 -> bf16 ----------------

__global__ void k_cvt_bf16(const float* __restrict__ src, bf16* __restrict__ dst, int n) {
  int i = blockIdx.x * blockDim.x + threadIdx.x;
  if (i < n) dst[i] = (bf16)src[i];
}

// ---------------- K1: context = attn @ enc_out ; also writes x[:, :E] bf16 ----------------

__global__ void k_context(const float* __restrict__ attn, const float* __restrict__ enc,
                          float* __restrict__ ctx, bf16* __restrict__ xbf) {
  int b = blockIdx.x;
  int e = threadIdx.x;  // 256 threads: cols e and e+256
  const float* er = enc + (size_t)b * S_ * E_;
  const float* ar = attn + (size_t)b * S_;
  float a0 = 0.f, a1 = 0.f;
  for (int s = 0; s < S_; ++s) {
    float w = ar[s];
    a0 += w * er[(size_t)s * E_ + e];
    a1 += w * er[(size_t)s * E_ + e + 256];
  }
  ctx[(size_t)b * E_ + e] = a0;
  ctx[(size_t)b * E_ + e + 256] = a1;
  xbf[(size_t)b * XK_ + e] = (bf16)a0;
  xbf[(size_t)b * XK_ + e + 256] = (bf16)a1;
}

// ---------------- K2: embedding gather ; writes x[:, E:] bf16 ----------------

__global__ void k_embed(const int* __restrict__ dec, const float* __restrict__ tab,
                        float* __restrict__ emb, bf16* __restrict__ xbf) {
  int b = blockIdx.x;
  int j = threadIdx.x;  // 128
  float v = tab[(size_t)dec[b] * EMB_ + j];
  emb[(size_t)b * EMB_ + j] = v;
  xbf[(size_t)b * XK_ + E_ + j] = (bf16)v;
}

// ---------------- K3: gates = x @ W_ih.T + h0 @ W_hh.T + biases (bf16 WMMA) ----------------
// grid (32 m-tiles, 16 n-groups of 64), block 128 (4 waves, 1 n-tile each)

__global__ void k_gates(const bf16* __restrict__ xbf, const bf16* __restrict__ h0bf,
                        const bf16* __restrict__ wih, const bf16* __restrict__ whh,
                        const float* __restrict__ bih, const float* __restrict__ bhh,
                        float* __restrict__ gates) {
  __shared__ bf16 lx[16 * XK_];
  __shared__ bf16 lh[16 * H_];
  int m0 = blockIdx.x * 16;
  int n0 = blockIdx.y * 64;
  int tid = threadIdx.x;
  for (int i = tid; i < 16 * XK_; i += 128)
    lx[i] = xbf[(size_t)(m0 + i / XK_) * XK_ + (i % XK_)];
  for (int i = tid; i < 16 * H_; i += 128)
    lh[i] = h0bf[(size_t)(m0 + i / H_) * H_ + (i % H_)];
  __syncthreads();

  int w = tid >> 5, lane = tid & 31, half = lane >> 4, nl = lane & 15;
  int nb = n0 + w * 16;
  v8f acc = {};
  for (int k0 = 0; k0 < XK_; k0 += 32) {
    v16bf a = frag_a(lx, XK_, 0, k0);
    v16bf bb = frag_b(wih, XK_, nb, k0);
    acc = wmma_bf16(a, bb, acc);
  }
  for (int k0 = 0; k0 < H_; k0 += 32) {
    v16bf a = frag_a(lh, H_, 0, k0);
    v16bf bb = frag_b(whh, H_, nb, k0);
    acc = wmma_bf16(a, bb, acc);
  }
  int n = nb + nl;
  float bias = bih[n] + bhh[n];
#pragma unroll
  for (int r = 0; r < 8; ++r) {
    int m = m0 + r + 8 * half;
    gates[(size_t)m * NG_ + n] = acc[r] + bias;
  }
}

// ---------------- K4: LSTM pointwise -> h_t, c_t ----------------

__global__ void k_lstm(const float* __restrict__ gates, const float* __restrict__ c0,
                       float* __restrict__ h_out, float* __restrict__ c_out,
                       bf16* __restrict__ hbf) {
  int idx = blockIdx.x * blockDim.x + threadIdx.x;  // B*H
  int b = idx / H_, h = idx % H_;
  const float* g = gates + (size_t)b * NG_;
  float gi = g[h], gf = g[h + H_], gg = g[h + 2 * H_], go = g[h + 3 * H_];
  float c = sigf(gf) * c0[idx] + sigf(gi) * tanhf(gg);
  float ht = sigf(go) * tanhf(c);
  c_out[idx] = c;
  h_out[idx] = ht;
  hbf[idx] = (bf16)ht;
}

// ---------------- K5a: ws_app = h_t @ aws_W.T + aws_b ----------------

__global__ void k_wsapp(const float* __restrict__ h, const float* __restrict__ awsW,
                        const float* __restrict__ awsb, float* __restrict__ wsapp) {
  int idx = blockIdx.x * blockDim.x + threadIdx.x;  // B*H
  int b = idx / H_, j = idx % H_;
  const float* hr = h + (size_t)b * H_;
  const float* wr = awsW + (size_t)j * H_;
  float a = awsb[j];
  for (int k = 0; k < H_; ++k) a += hr[k] * wr[k];
  wsapp[idx] = a;
}

// ---------------- K5b: p_gen ----------------

__global__ void k_pgen(const float* __restrict__ ctx, const float* __restrict__ h,
                       const float* __restrict__ emb, const float* __restrict__ wh,
                       const float* __restrict__ wsv, const float* __restrict__ wx,
                       float* __restrict__ pgen) {
  int b = blockIdx.x, t = threadIdx.x;  // 128 threads
  float a = 0.f;
  for (int i = t; i < E_; i += 128) a += ctx[(size_t)b * E_ + i] * wh[i];
  for (int i = t; i < H_; i += 128) a += h[(size_t)b * H_ + i] * wsv[i];
  for (int i = t; i < EMB_; i += 128) a += emb[(size_t)b * EMB_ + i] * wx[i];
  __shared__ float red[128];
  red[t] = a;
  __syncthreads();
  for (int s = 64; s > 0; s >>= 1) {
    if (t < s) red[t] += red[t + s];
    __syncthreads();
  }
  if (t == 0) pgen[b] = sigf(red[0]);
}

// ---------------- K6: energy kernel (dominant): bf16 WMMA over enc_out tiles ----------------
// energy[b,s] = sum_h av[h] * tanh( (enc_out[b,s,:] @ awh_W[h,:]) + awh_b[h]
//                                   + ws_app[b,h] + coverage[b,s]*awc[h] )
// grid (25 s-tiles, 512 b), block 128 (4 waves x 4 n-tiles = all H=256)

__global__ void k_energy(const float* __restrict__ enc, const bf16* __restrict__ awhbf,
                         const float* __restrict__ awhb, const float* __restrict__ wsapp,
                         const float* __restrict__ cov, const float* __restrict__ awc,
                         const float* __restrict__ av, float* __restrict__ energy) {
  __shared__ bf16 la[16 * E_];
  __shared__ float red[4][16];
  int st = blockIdx.x * 16;
  int b = blockIdx.y;
  int tid = threadIdx.x;
  const float* et = enc + ((size_t)b * S_ + st) * E_;
  for (int i = tid; i < 16 * E_; i += 128) la[i] = (bf16)et[i];
  __syncthreads();

  int w = tid >> 5, lane = tid & 31, half = lane >> 4, nl = lane & 15;
  v8f acc[4] = {};
  for (int k0 = 0; k0 < E_; k0 += 32) {
    v16bf a = frag_a(la, E_, 0, k0);
#pragma unroll
    for (int t = 0; t < 4; ++t) {
      v16bf bb = frag_b(awhbf, E_, w * 64 + t * 16, k0);
      acc[t] = wmma_bf16(a, bb, acc[t]);
    }
  }

  float part[8];
#pragma unroll
  for (int r = 0; r < 8; ++r) part[r] = 0.f;
#pragma unroll
  for (int t = 0; t < 4; ++t) {
    int n = w * 64 + t * 16 + nl;
    float add_n = wsapp[(size_t)b * H_ + n] + awhb[n];
    float awcn = awc[n], avn = av[n];
#pragma unroll
    for (int r = 0; r < 8; ++r) {
      int s = st + r + 8 * half;
      float val = acc[t][r] + add_n + cov[(size_t)b * S_ + s] * awcn;
      part[r] += tanhf(val) * avn;
    }
  }
  // reduce across the 16 lanes of each half (they hold the 16 h-columns of a row)
#pragma unroll
  for (int r = 0; r < 8; ++r) {
#pragma unroll
    for (int off = 1; off < 16; off <<= 1) part[r] += __shfl_xor(part[r], off, 32);
  }
  if (nl == 0) {
#pragma unroll
    for (int r = 0; r < 8; ++r) red[w][r + 8 * half] = part[r];
  }
  __syncthreads();
  if (tid < 16) {
    energy[(size_t)b * S_ + st + tid] =
        red[0][tid] + red[1][tid] + red[2][tid] + red[3][tid];
  }
}

// ---------------- K7: attention softmax + coverage update + per-b loss ----------------

__global__ void k_attn(const float* __restrict__ energy, const float* __restrict__ cov,
                       float* __restrict__ attn_out, float* __restrict__ cov_out,
                       float* __restrict__ loss_part) {
  int b = blockIdx.x, t = threadIdx.x;  // 128 threads
  __shared__ float red[128];
  const float* er = energy + (size_t)b * S_;
  float mx = -1e30f;
  for (int s = t; s < S_; s += 128) mx = fmaxf(mx, er[s]);
  red[t] = mx; __syncthreads();
  for (int s = 64; s > 0; s >>= 1) { if (t < s) red[t] = fmaxf(red[t], red[t + s]); __syncthreads(); }
  mx = red[0]; __syncthreads();
  float sm = 0.f;
  for (int s = t; s < S_; s += 128) sm += __expf(er[s] - mx);
  red[t] = sm; __syncthreads();
  for (int s = 64; s > 0; s >>= 1) { if (t < s) red[t] += red[t + s]; __syncthreads(); }
  float inv = 1.f / red[0]; __syncthreads();
  float lloc = 0.f;
  for (int s = t; s < S_; s += 128) {
    float a = __expf(er[s] - mx) * inv;
    float c = cov[(size_t)b * S_ + s];
    attn_out[(size_t)b * S_ + s] = a;
    cov_out[(size_t)b * S_ + s] = c + a;
    lloc += fminf(a, c);
  }
  red[t] = lloc; __syncthreads();
  for (int s = 64; s > 0; s >>= 1) { if (t < s) red[t] += red[t + s]; __syncthreads(); }
  if (t == 0) loss_part[b] = red[0];
}

// ---------------- K8: reduce 512 per-b losses -> scalar ----------------

__global__ void k_loss(const float* __restrict__ lp, float* __restrict__ out_loss) {
  __shared__ float red[256];
  int t = threadIdx.x;
  red[t] = lp[t] + lp[t + 256];
  __syncthreads();
  for (int s = 128; s > 0; s >>= 1) { if (t < s) red[t] += red[t + s]; __syncthreads(); }
  if (t == 0) *out_loss = red[0];
}

// ---------------- K9: logits = h_t @ v_W.T + v_b (bf16 WMMA), raw into d_out ----------------
// grid (196 n-blocks of 256, 32 m-tiles), block 128 (4 waves x 4 n-tiles)

__global__ void k_logits(const bf16* __restrict__ hbf, const bf16* __restrict__ vwbf,
                         const float* __restrict__ vb, float* __restrict__ out) {
  int m0 = blockIdx.y * 16;
  int tid = threadIdx.x;
  int w = tid >> 5, lane = tid & 31, half = lane >> 4, nl = lane & 15;
  int nbase = blockIdx.x * 256 + w * 64;
  v8f acc[4] = {};
  for (int k0 = 0; k0 < H_; k0 += 32) {
    v16bf a = frag_a(hbf, H_, m0, k0);
#pragma unroll
    for (int t = 0; t < 4; ++t) {
      if (nbase + t * 16 < V_) {  // wave-uniform predicate: EXEC stays all-ones
        v16bf bb = frag_b(vwbf, H_, nbase + t * 16, k0);
        acc[t] = wmma_bf16(a, bb, acc[t]);
      }
    }
  }
#pragma unroll
  for (int t = 0; t < 4; ++t) {
    if (nbase + t * 16 >= V_) continue;
    int n = nbase + t * 16 + nl;
    float bias = vb[n];
#pragma unroll
    for (int r = 0; r < 8; ++r) {
      int m = m0 + r + 8 * half;
      out[(size_t)m * V_ + n] = acc[t][r] + bias;
    }
  }
}

// ---------------- K10: vocab softmax (online) * p_gen, in place ----------------

__global__ void k_vsoftmax(float* __restrict__ logits, const float* __restrict__ pgen) {
  int b = blockIdx.x, t = threadIdx.x;  // 256 threads
  float* row = logits + (size_t)b * V_;
  float mx = -1e30f, sm = 0.f;
  for (int i = t; i < V_; i += 256) {
    float x = row[i];
    float nm = fmaxf(mx, x);
    sm = sm * __expf(mx - nm) + __expf(x - nm);
    mx = nm;
  }
  __shared__ float smx[256], ssm[256];
  smx[t] = mx; ssm[t] = sm;
  __syncthreads();
  for (int s = 128; s > 0; s >>= 1) {
    if (t < s) {
      float m2 = smx[t + s], s2 = ssm[t + s];
      float nm = fmaxf(smx[t], m2);
      ssm[t] = ssm[t] * __expf(smx[t] - nm) + s2 * __expf(m2 - nm);
      smx[t] = nm;
    }
    __syncthreads();
  }
  float M = smx[0], scale = pgen[b] / ssm[0];
  for (int i = t; i < V_; i += 256) row[i] = __expf(row[i] - M) * scale;
}

// ---------------- K11: scatter .set semantics: out[b, idx] = pv + (1-pg)*attn ----------------
// gather originals first, then write ascending (last-wins) -> deterministic, no atomics

__global__ void k_scatter(const float* __restrict__ attn, const int* __restrict__ encin,
                          const float* __restrict__ pgen, float* __restrict__ out,
                          float* __restrict__ tmp) {
  int b = blockIdx.x * blockDim.x + threadIdx.x;
  if (b >= B_) return;
  float g = 1.f - pgen[b];
  float* row = out + (size_t)b * V_;
  const int* ir = encin + (size_t)b * S_;
  const float* ar = attn + (size_t)b * S_;
  float* tr = tmp + (size_t)b * S_;
  for (int s = 0; s < S_; ++s) tr[s] = row[ir[s]];
  for (int s = 0; s < S_; ++s) row[ir[s]] = tr[s] + g * ar[s];
}

// ---------------- host launcher ----------------

extern "C" void kernel_launch(void* const* d_in, const int* in_sizes, int n_in,
                              void* d_out, int out_size, void* d_ws, size_t ws_size,
                              hipStream_t stream) {
  const float* coverage = (const float*)d_in[0];
  const float* enc_out  = (const float*)d_in[1];
  const float* h0       = (const float*)d_in[2];
  const float* c0       = (const float*)d_in[3];
  const float* attn     = (const float*)d_in[4];
  const int*   dec_in   = (const int*)d_in[5];
  const int*   enc_in   = (const int*)d_in[6];
  const float* emb_tab  = (const float*)d_in[7];
  const float* W_ih     = (const float*)d_in[8];
  const float* W_hh     = (const float*)d_in[9];
  const float* b_ih     = (const float*)d_in[10];
  const float* b_hh     = (const float*)d_in[11];
  const float* awh_W    = (const float*)d_in[12];
  const float* awh_b    = (const float*)d_in[13];
  const float* aws_W    = (const float*)d_in[14];
  const float* aws_b    = (const float*)d_in[15];
  const float* awc      = (const float*)d_in[16];
  const float* av       = (const float*)d_in[17];
  const float* wh       = (const float*)d_in[18];
  const float* wsv      = (const float*)d_in[19];
  const float* wx       = (const float*)d_in[20];
  const float* v_W      = (const float*)d_in[21];
  const float* v_b      = (const float*)d_in[22];

  // d_out layout (floats): output(B*V) | new_coverage(B*S) | h_t(B*H) | c_t(B*H) | new_attn(B*S) | loss(1)
  float* out_p    = (float*)d_out;
  float* out_cov  = out_p + (size_t)B_ * V_;
  float* out_h    = out_cov + (size_t)B_ * S_;
  float* out_c    = out_h + (size_t)B_ * H_;
  float* out_attn = out_c + (size_t)B_ * H_;
  float* out_loss = out_attn + (size_t)B_ * S_;

  // workspace layout (bytes, all chunks 256B-multiples)
  char* ws = (char*)d_ws;
  size_t off = 0;
  float* ctx    = (float*)(ws + off); off += (size_t)B_ * E_ * 4;      // 1,048,576
  float* emb    = (float*)(ws + off); off += (size_t)B_ * EMB_ * 4;    //   262,144
  bf16*  xbf    = (bf16*)(ws + off);  off += (size_t)B_ * XK_ * 2;     //   655,360
  bf16*  h0bf   = (bf16*)(ws + off);  off += (size_t)B_ * H_ * 2;      //   262,144
  float* gates  = (float*)(ws + off); off += (size_t)B_ * NG_ * 4;     // 2,097,152
  bf16*  hbf    = (bf16*)(ws + off);  off += (size_t)B_ * H_ * 2;      //   262,144
  float* wsapp  = (float*)(ws + off); off += (size_t)B_ * H_ * 4;      //   524,288
  float* pgen   = (float*)(ws + off); off += 2048;
  float* energy = (float*)(ws + off); off += (size_t)B_ * S_ * 4;      //   819,200
  float* losspt = (float*)(ws + off); off += 2048;
  bf16*  wihbf  = (bf16*)(ws + off);  off += (size_t)NG_ * XK_ * 2;    // 1,310,720
  bf16*  whhbf  = (bf16*)(ws + off);  off += (size_t)NG_ * H_ * 2;     //   524,288
  bf16*  awhbf  = (bf16*)(ws + off);  off += (size_t)H_ * E_ * 2;      //   262,144
  bf16*  vwbf   = (bf16*)(ws + off);  off += (size_t)V_ * H_ * 2;      // 25,600,000
  (void)off; (void)ws_size; (void)n_in; (void)in_sizes; (void)out_size;

  // weight / state conversions to bf16
  k_cvt_bf16<<<(NG_ * XK_ + 255) / 256, 256, 0, stream>>>(W_ih, wihbf, NG_ * XK_);
  k_cvt_bf16<<<(NG_ * H_ + 255) / 256, 256, 0, stream>>>(W_hh, whhbf, NG_ * H_);
  k_cvt_bf16<<<(H_ * E_ + 255) / 256, 256, 0, stream>>>(awh_W, awhbf, H_ * E_);
  k_cvt_bf16<<<(V_ * H_ + 255) / 256, 256, 0, stream>>>(v_W, vwbf, V_ * H_);
  k_cvt_bf16<<<(B_ * H_ + 255) / 256, 256, 0, stream>>>(h0, h0bf, B_ * H_);

  // pipeline
  k_context<<<B_, 256, 0, stream>>>(attn, enc_out, ctx, xbf);
  k_embed<<<B_, 128, 0, stream>>>(dec_in, emb_tab, emb, xbf);
  k_gates<<<dim3(B_ / 16, NG_ / 64), 128, 0, stream>>>(xbf, h0bf, wihbf, whhbf, b_ih, b_hh, gates);
  k_lstm<<<(B_ * H_) / 256, 256, 0, stream>>>(gates, c0, out_h, out_c, hbf);
  k_wsapp<<<(B_ * H_) / 256, 256, 0, stream>>>(out_h, aws_W, aws_b, wsapp);
  k_pgen<<<B_, 128, 0, stream>>>(ctx, out_h, emb, wh, wsv, wx, pgen);
  k_energy<<<dim3(S_ / 16, B_), 128, 0, stream>>>(enc_out, awhbf, awh_b, wsapp, coverage,
                                                  awc, av, energy);
  k_attn<<<B_, 128, 0, stream>>>(energy, coverage, out_attn, out_cov, losspt);
  k_loss<<<1, 256, 0, stream>>>(losspt, out_loss);
  k_logits<<<dim3((V_ / 16 + 15) / 16, B_ / 16), 128, 0, stream>>>(hbf, vwbf, v_b, out_p);
  k_vsoftmax<<<B_, 256, 0, stream>>>(out_p, pgen);
  // reuse energy buffer as scatter gather-scratch (no longer needed)
  k_scatter<<<2, 256, 0, stream>>>(out_attn, enc_in, pgen, out_p, energy);
}
